// MwAN_56212531970502
// MI455X (gfx1250) — matrix operand order repaired and checked by reading, not compiled, not run on hardware
//
#include <hip/hip_runtime.h>
#include <hip/hip_bf16.h>
#include <math.h>

// ---------------------------------------------------------------------------
// MwAN forward (training loss) for MI455X / gfx1250, wave32 + WMMA f16.
// All GEMM-like work goes through v_wmma_f32_16x16x32_f16 (f32 accumulate).
// Every WMMA operand (A and B) is loaded as contiguous float4 -> b128:
// all B matrices are staged transposed (N-major, zero-padded) in workspace.
// ---------------------------------------------------------------------------

typedef __attribute__((ext_vector_type(16))) _Float16 v16h;
typedef __attribute__((ext_vector_type(8)))  float    v8f;

#define BB     4
#define QL     64
#define PL     256
#define ALEN   16
#define NANS   3
#define EE     300
#define EPAD   320   // 300 padded to multiple of 32 (K dimension)
#define EPAD16 304   // 300 padded to multiple of 16 (N dimension)
#define HH     128
#define D8     1024
#define FF     512
#define NHEADS 8

// ---- parameter index map (setup_inputs dict order, nested insertion order) --
enum ParamIdx {
  IN_QUERY = 0, IN_PASSAGE = 1, IN_ANSWER = 2, IN_ISTRAIN = 3,
  P_EMB = 4,
  P_PROJ_Q_W = 5, P_PROJ_Q_B = 6,
  P_PROJ_P_W = 7, P_PROJ_P_B = 8,
  P_PROJ_A_W = 9, P_PROJ_A_B = 10,
  P_CONV0 = 11,                 // 2 x {ln_g, ln_b, dw, dw_b, pw, pw_b}
  P_CONV_LN_G = 23, P_CONV_LN_B = 24,
  P_ENC0 = 25,                  // 3 x {ln1_g, ln1_b, ln2_g, ln2_b, w1, b1, w2, b2}
  P_ENC_LN_G = 49, P_ENC_LN_B = 50,
  P_A_PROJ_W = 51, P_A_PROJ_B = 52, P_A_ATT = 53,
  P_WC1 = 54, P_WC2 = 55, P_VC = 56,
  P_WB = 57,
  P_WD = 58, P_VD = 59,
  P_WM = 60, P_VM = 61,
  P_WS = 62, P_VS = 63,
  P_WQA1 = 64, P_WQA2 = 65, P_WQA3 = 66,
  P_ACONV0 = 67,                // 2 x {ln_g, ln_b, dw, dw_b, pw, pw_b}
  P_ACONV_LN_G = 79, P_ACONV_LN_B = 80,
  P_AENC0 = 81,                 // 4 x {ln1..w2,b2}
  P_AENC_LN_G = 113, P_AENC_LN_B = 114,
  P_WQ = 115, P_VQ = 116,
  P_WP1 = 117, P_WP2 = 118, P_VP = 119,
  P_PRED = 120
};

// epilogue flags for the generic WMMA GEMM
enum { EP_BIAS = 1, EP_RELU = 2, EP_RESID = 4, EP_ADDROW = 8, EP_ADDCOL = 16,
       EP_LEAKY = 32 };

// ---------------------------------------------------------------------------
// device helpers
// ---------------------------------------------------------------------------
__device__ __forceinline__ float block_reduce(float v, float* sm, int opmax) {
  int t = threadIdx.x;
  sm[t] = v;
  __syncthreads();
  for (int s = blockDim.x >> 1; s > 0; s >>= 1) {
    if (t < s) sm[t] = opmax ? fmaxf(sm[t], sm[t + s]) : (sm[t] + sm[t + s]);
    __syncthreads();
  }
  float r = sm[0];
  __syncthreads();
  return r;
}

__device__ __forceinline__ v16h cvt_frag(float4 x0, float4 x1, float4 x2, float4 x3) {
  v16h a;
  a[0]  = (_Float16)x0.x; a[1]  = (_Float16)x0.y; a[2]  = (_Float16)x0.z; a[3]  = (_Float16)x0.w;
  a[4]  = (_Float16)x1.x; a[5]  = (_Float16)x1.y; a[6]  = (_Float16)x1.z; a[7]  = (_Float16)x1.w;
  a[8]  = (_Float16)x2.x; a[9]  = (_Float16)x2.y; a[10] = (_Float16)x2.z; a[11] = (_Float16)x2.w;
  a[12] = (_Float16)x3.x; a[13] = (_Float16)x3.y; a[14] = (_Float16)x3.z; a[15] = (_Float16)x3.w;
  return a;
}

// A-matrix 16x32 f16 lane layout: lane holds K = off..off+7 and off+16..off+23
__device__ __forceinline__ v16h load_a_frag(const float* row, int off) {
  const float4* p0 = (const float4*)(row + off);
  const float4* p1 = (const float4*)(row + off + 16);
  return cvt_frag(p0[0], p0[1], p1[0], p1[1]);
}
// B fragment from N-major (transposed) storage: K = off..off+7 and off+16..off+23
__device__ __forceinline__ v16h load_bt_frag(const float* row, int off) {
  const float4* p0 = (const float4*)(row + off);
  const float4* p1 = (const float4*)(row + off + 16);
  return cvt_frag(p0[0], p0[1], p1[0], p1[1]);
}

__device__ __forceinline__ float4 f4op(float4 a, float4 b, int mode) {
  float4 r;
  r.x = mode ? (a.x - b.x) : (a.x * b.x);
  r.y = mode ? (a.y - b.y) : (a.y * b.y);
  r.z = mode ? (a.z - b.z) : (a.z * b.z);
  r.w = mode ? (a.w - b.w) : (a.w * b.w);
  return r;
}

// ---------------------------------------------------------------------------
// generic tiled WMMA GEMM: C[M,N] = A[M,K] * Bt[N,K]^T  (+epilogue)
// one wave (32 threads) per 16x16 output tile; f32->f16 convert on load,
// f32 accumulation. Inner loop branch-free: K multiple of 32, A rows and Bt
// rows readable up to the padded tile edge (WS slack / zero-padded weights).
// Only the epilogue store is bound-checked.
// ---------------------------------------------------------------------------
__global__ void wmma_gemm_kernel(const float* __restrict__ A,
                                 const float* __restrict__ Bt,
                                 const float* __restrict__ bias,
                                 const float* __restrict__ resid,
                                 const float* __restrict__ rowv,
                                 const float* __restrict__ colv,
                                 float* __restrict__ C,
                                 int M, int N, int K, int lda, int ldb, int ldc,
                                 int flags, long sA, long sB, long sC,
                                 long sRow, long sCol) {
  const int tn = blockIdx.x * 16;
  const int tm = blockIdx.y * 16;
  const int b  = blockIdx.z;
  const int lane  = threadIdx.x;
  const int khalf = (lane >> 4) * 8;     // A-matrix lane layout
  const int kbase = (lane >> 4) * 8;     // B-matrix lane layout

  const float* arow = A + (long)b * sA + (long)(tm + (lane & 15)) * lda;
  const float* brow = Bt + (long)b * sB + (long)(tn + (lane & 15)) * ldb;
  v8f acc = {};
  for (int k0 = 0; k0 < K; k0 += 32) {
    __builtin_prefetch(arow + k0 + 64, 0, 1);  // speculative, non-faulting
    v16h af = load_a_frag(arow, k0 + khalf);
    v16h bf = load_bt_frag(brow, k0 + kbase);
    acc = __builtin_amdgcn_wmma_f32_16x16x32_f16(false, af, false, bf,
                                                 (short)0, acc, false, false);
  }
  // epilogue: C/D layout — lane half selects row+8, col = lane&15
  float* Cb = C + (long)b * sC;
  const int n = tn + (lane & 15);
  const int rbase = tm + ((lane >> 4) ? 8 : 0);
#pragma unroll
  for (int r = 0; r < 8; ++r) {
    int m = rbase + r;
    if (m < M && n < N) {
      float v = acc[r];
      if (flags & EP_BIAS)   v += bias[n];
      if (flags & EP_ADDROW) v += rowv[(long)b * sRow + n];
      if (flags & EP_ADDCOL) v += colv[(long)b * sCol + m];
      if (flags & EP_RELU)   v = fmaxf(v, 0.0f);
      if (flags & EP_LEAKY)  v = (v > 0.0f) ? v : 0.01f * v;
      if (flags & EP_RESID)  v += resid[(long)b * sC + (long)m * ldc + n];
      Cb[(long)m * ldc + n] = v;
    }
  }
}

// ---------------------------------------------------------------------------
// fused multiway score: s[b,q,p] = sum_n tanh( ((hp[p] op hq[q]) @ W)[n] )*v[n]
// op: mode 0 -> elementwise product, mode 1 -> difference.
// Wt is W transposed [H,H] so B fragments are contiguous b128 loads.
// One wave per (b, q, 16-wide p-tile); the 16x128 A-tile is formed in
// registers (the rank-4 tensor never hits HBM).
// ---------------------------------------------------------------------------
__global__ void mw_tanh_score_kernel(const float* __restrict__ hp,
                                     const float* __restrict__ hq,
                                     const float* __restrict__ Wt,
                                     const float* __restrict__ v,
                                     float* __restrict__ s,
                                     int Pn, int Qn, int H, int mode) {
  const int b = blockIdx.z, q = blockIdx.y, p0 = blockIdx.x * 16;
  const int lane = threadIdx.x;
  const float* hqr = hq + ((long)b * Qn + q) * H;
  const float* hprow = hp + ((long)b * Pn + p0 + (lane & 15)) * H;
  const int khalf = (lane >> 4) * 8;
  const int ncol  = lane & 15;
  const int kbase = (lane >> 4) * 8;

  float part[8];
#pragma unroll
  for (int r = 0; r < 8; ++r) part[r] = 0.0f;

  for (int n0 = 0; n0 < H; n0 += 16) {
    const float* wrow = Wt + (long)(n0 + ncol) * H;
    v8f acc = {};
    for (int k0 = 0; k0 < H; k0 += 32) {
      const float4* pp0 = (const float4*)(hprow + k0 + khalf);
      const float4* pp1 = (const float4*)(hprow + k0 + khalf + 16);
      const float4* qq0 = (const float4*)(hqr + k0 + khalf);
      const float4* qq1 = (const float4*)(hqr + k0 + khalf + 16);
      v16h af = cvt_frag(f4op(pp0[0], qq0[0], mode), f4op(pp0[1], qq0[1], mode),
                         f4op(pp1[0], qq1[0], mode), f4op(pp1[1], qq1[1], mode));
      v16h bf = load_bt_frag(wrow, k0 + kbase);
      acc = __builtin_amdgcn_wmma_f32_16x16x32_f16(false, af, false, bf,
                                                   (short)0, acc, false, false);
    }
    float vn = v[n0 + ncol];
#pragma unroll
    for (int r = 0; r < 8; ++r) part[r] += tanhf(acc[r]) * vn;
  }
  // sum over the 16 columns held by the 16 lanes of each half-wave
#pragma unroll
  for (int r = 0; r < 8; ++r) {
    float x = part[r];
    x += __shfl_xor(x, 1, 32);
    x += __shfl_xor(x, 2, 32);
    x += __shfl_xor(x, 4, 32);
    x += __shfl_xor(x, 8, 32);
    part[r] = x;
  }
  if ((lane & 15) == 0) {
    int rowoff = (lane >> 4) * 8;
#pragma unroll
    for (int r = 0; r < 8; ++r)
      s[((long)b * Qn + q) * Pn + p0 + rowoff + r] = part[r];
  }
}

// transpose + zero-pad: dst[n, k] = src[k, n]; dst is [drows(grid), dcols]
__global__ void transpose_pad_kernel(const float* __restrict__ src,
                                     float* __restrict__ dst,
                                     int rows, int cols, int dcols) {
  int n = blockIdx.x;  // dst row == src column
  for (int k = threadIdx.x; k < dcols; k += blockDim.x)
    dst[(long)n * dcols + k] =
        (k < rows && n < cols) ? src[(long)k * cols + n] : 0.0f;
}

// batched transpose: src [b, rows, cols] -> dst [b, cols, rows]
__global__ void transpose_batch_kernel(const float* __restrict__ src,
                                       float* __restrict__ dst,
                                       int rows, int cols) {
  int b = blockIdx.y, c = blockIdx.x;
  for (int r = threadIdx.x; r < rows; r += blockDim.x)
    dst[((long)b * cols + c) * rows + r] = src[((long)b * rows + r) * cols + c];
}

// embedding gather + sinusoidal positional encoding, row stride ldo (zero tail)
__global__ void embed_posenc_kernel(const int* __restrict__ tok,
                                    const float* __restrict__ emb,
                                    float* __restrict__ out, int L, int E, int ldo) {
  long row = blockIdx.x;
  int l = (int)(row % L);
  int id = tok[row];
  for (int e = threadIdx.x; e < ldo; e += blockDim.x) {
    float val = 0.0f;
    if (e < E) {
      int i2 = e & ~1;
      float div = __expf(-logf(10000.0f) * (float)i2 / (float)E);
      float ang = (float)l * div;
      float pe = (e & 1) ? __cosf(ang) : __sinf(ang);
      val = emb[(long)id * E + e] + pe;
    }
    out[row * (long)ldo + e] = val;
  }
}

__global__ void layernorm_kernel(const float* __restrict__ x,
                                 const float* __restrict__ g,
                                 const float* __restrict__ bt,
                                 float* __restrict__ out, int D) {
  __shared__ float red[256];
  long row = blockIdx.x;
  const float* xr = x + row * D;
  float s = 0.0f, ss = 0.0f;
  for (int i = threadIdx.x; i < D; i += blockDim.x) {
    float v = xr[i]; s += v; ss += v * v;
  }
  s = block_reduce(s, red, 0);
  ss = block_reduce(ss, red, 0);
  float m = s / D;
  float var = ss / D - m * m;
  float inv = rsqrtf(var + 1e-6f);
  for (int i = threadIdx.x; i < D; i += blockDim.x)
    out[row * D + i] = (xr[i] - m) * inv * g[i] + bt[i];
}

// depthwise 5-tap conv along L, SAME padding, + bias. w flat [5,1,C] -> t*C+c
__global__ void dwconv_kernel(const float* __restrict__ x,
                              const float* __restrict__ w,
                              const float* __restrict__ bias,
                              float* __restrict__ out, int L, int C) {
  int b = blockIdx.y, l = blockIdx.x;
  const float* xb = x + (long)b * L * C;
  for (int c = threadIdx.x; c < C; c += blockDim.x) {
    float acc = bias[c];
#pragma unroll
    for (int t = 0; t < 5; ++t) {
      int ll = l + t - 2;
      if (ll >= 0 && ll < L) acc += xb[(long)ll * C + c] * w[t * C + c];
    }
    out[((long)b * L + l) * C + c] = acc;
  }
}

// projection-free multi-head self attention: out = resid + mha(xln)
__global__ void mha_kernel(const float* __restrict__ xln,
                           const float* __restrict__ resid,
                           const int* __restrict__ tokens,
                           float* __restrict__ out, int L, int D, int heads) {
  __shared__ float sc[256];
  __shared__ float red[256];
  int b = blockIdx.y, q = blockIdx.x, t = threadIdx.x;
  int d = D / heads;
  float scale = rsqrtf((float)d);
  const float* xb = xln + (long)b * L * D;
  for (int h = 0; h < heads; ++h) {
    const float* xh = xb + h * d;
    float sv = -1e30f;
    if (t < L) {
      const float* qr = xh + (long)q * D;
      const float* kr = xh + (long)t * D;
      float acc = 0.0f;
      for (int j = 0; j < d; ++j) acc += qr[j] * kr[j];
      sv = acc * scale;
      if (tokens[b * L + t] == 0) sv = -1e9f;
    }
    float mx = block_reduce(sv, red, 1);
    float e = (t < L) ? __expf(sv - mx) : 0.0f;
    float sum = block_reduce(e, red, 0);
    sc[t] = e / sum;
    __syncthreads();
    for (int j = t; j < d; j += blockDim.x) {
      float acc = 0.0f;
      for (int k = 0; k < L; ++k) acc += sc[k] * xh[(long)k * D + j];
      long oi = ((long)b * L + q) * D + h * d + j;
      out[oi] = resid[oi] + acc;
    }
    __syncthreads();
  }
}

// softmax over last dim of s, then out[b,q,:] = sum_p a[p] * V[b,p,:]
__global__ void attn_pool_kernel(const float* __restrict__ s,
                                 const float* __restrict__ V,
                                 float* __restrict__ out,
                                 int Q, int P, int H, int ldo, long sOutB) {
  __shared__ float sc[256];
  __shared__ float red[256];
  int b = blockIdx.y, q = blockIdx.x, t = threadIdx.x;
  const float* sr = s + ((long)b * Q + q) * P;
  float sv = (t < P) ? sr[t] : -1e30f;
  float mx = block_reduce(sv, red, 1);
  float e = (t < P) ? __expf(sv - mx) : 0.0f;
  float sum = block_reduce(e, red, 0);
  sc[t] = e / sum;
  __syncthreads();
  const float* Vb = V + (long)b * P * H;
  for (int j = t; j < H; j += blockDim.x) {
    float acc = 0.0f;
    for (int k = 0; k < P; ++k) acc += sc[k] * Vb[(long)k * H + j];
    out[(long)b * sOutB + (long)q * ldo + j] = acc;
  }
}

// s[b,q,p] = sum_h tanh(c1[b,p,h] + c2[b,q,h]) * v[h]
__global__ void addtanh_score_kernel(const float* __restrict__ c1,
                                     const float* __restrict__ c2,
                                     const float* __restrict__ v,
                                     float* __restrict__ s, int Q, int P, int H) {
  int b = blockIdx.y, q = blockIdx.x;
  const float* c2r = c2 + ((long)b * Q + q) * H;
  for (int p = threadIdx.x; p < P; p += blockDim.x) {
    const float* c1r = c1 + ((long)b * P + p) * H;
    float acc = 0.0f;
    for (int h = 0; h < H; ++h) acc += tanhf(c1r[h] + c2r[h]) * v[h];
    s[((long)b * Q + q) * P + p] = acc;
  }
}

__global__ void softmax_lastdim_kernel(float* __restrict__ dst,
                                       const float* __restrict__ src, int n) {
  __shared__ float red[256];
  long row = blockIdx.x;
  int t = threadIdx.x;
  float v = (t < n) ? src[row * n + t] : -1e30f;
  float mx = block_reduce(v, red, 1);
  float e = (t < n) ? __expf(v - mx) : 0.0f;
  float sum = block_reduce(e, red, 0);
  if (t < n) dst[row * n + t] = e / sum;
}

// softmax over dim1 (q) of [B,Q,P]
__global__ void softmax_dim1_kernel(float* __restrict__ dst,
                                    const float* __restrict__ src, int Q, int P) {
  __shared__ float red[64];
  int b = blockIdx.y, p = blockIdx.x, t = threadIdx.x;
  long base = (long)b * Q * P + p;
  float v = (t < Q) ? src[base + (long)t * P] : -1e30f;
  float mx = block_reduce(v, red, 1);
  float e = (t < Q) ? __expf(v - mx) : 0.0f;
  float sum = block_reduce(e, red, 0);
  if (t < Q) dst[base + (long)t * P] = e / sum;
}

// s[row] = sum_i f(X[row,i]) * v[i], f = tanh optionally
__global__ void rowdot_kernel(const float* __restrict__ X,
                              const float* __restrict__ v,
                              float* __restrict__ s, int D, int ldx, int useTanh) {
  __shared__ float red[256];
  long row = blockIdx.x;
  float acc = 0.0f;
  for (int i = threadIdx.x; i < D; i += blockDim.x) {
    float x = X[row * (long)ldx + i];
    if (useTanh) x = tanhf(x);
    acc += x * v[i];
  }
  acc = block_reduce(acc, red, 0);
  if (threadIdx.x == 0) s[row] = acc;
}

// out[b,:] = sum_l softmax(s[b,:L])[l] * V[b,l,:]
__global__ void pool_rows_kernel(const float* __restrict__ s,
                                 const float* __restrict__ V,
                                 float* __restrict__ out, int L, int H) {
  __shared__ float sc[256];
  __shared__ float red[256];
  int b = blockIdx.x, t = threadIdx.x;
  float sv = (t < L) ? s[(long)b * L + t] : -1e30f;
  float mx = block_reduce(sv, red, 1);
  float e = (t < L) ? __expf(sv - mx) : 0.0f;
  float sum = block_reduce(e, red, 0);
  sc[t] = e / sum;
  __syncthreads();
  for (int j = t; j < H; j += blockDim.x) {
    float acc = 0.0f;
    for (int k = 0; k < L; ++k) acc += sc[k] * V[((long)b * L + k) * H + j];
    out[(long)b * H + j] = acc;
  }
}

__global__ void copy_slice_kernel(const float* __restrict__ src,
                                  float* __restrict__ dst, int H,
                                  int ldsrc, int lddst) {
  long row = blockIdx.x;
  for (int j = threadIdx.x; j < H; j += blockDim.x)
    dst[row * lddst + j] = src[row * ldsrc + j];
}

__global__ void zero_out_kernel(float* out) { out[0] = 0.0f; }

__global__ void loss_kernel(const float* __restrict__ a_emb,
                            const float* __restrict__ eo,
                            float* __restrict__ out, int na, int E, int Bt) {
  __shared__ float red[256];
  __shared__ float dots[8];
  int b = blockIdx.x, t = threadIdx.x;
  for (int k = 0; k < na; ++k) {
    float acc = 0.0f;
    for (int i = t; i < E; i += blockDim.x)
      acc += a_emb[((long)b * na + k) * E + i] * eo[(long)b * E + i];
    acc = block_reduce(acc, red, 0);
    if (t == 0) dots[k] = acc;
    __syncthreads();
  }
  if (t == 0) {
    float mx = dots[0];
    for (int k = 1; k < na; ++k) mx = fmaxf(mx, dots[k]);
    float sum = 0.0f;
    for (int k = 0; k < na; ++k) sum += __expf(dots[k] - mx);
    float p0 = __expf(dots[0] - mx) / sum;
    atomicAdd(out, -__logf(p0) / (float)Bt);
  }
}

// ---------------------------------------------------------------------------
// host helpers
// ---------------------------------------------------------------------------
static void launch_gemm(hipStream_t st, const float* A, const float* Bt,
                        const float* bias, const float* resid,
                        const float* rowv, const float* colv, float* C,
                        int M, int N, int K, int lda, int ldb, int ldc,
                        int flags, int batch, long sA, long sB, long sC,
                        long sRow, long sCol) {
  dim3 g((N + 15) / 16, (M + 15) / 16, batch);
  wmma_gemm_kernel<<<g, 32, 0, st>>>(A, Bt, bias, resid, rowv, colv, C,
                                     M, N, K, lda, ldb, ldc, flags,
                                     sA, sB, sC, sRow, sCol);
}
static void gemm1(hipStream_t st, const float* A, const float* Bt,
                  const float* bias, const float* resid, float* C,
                  int M, int N, int K, int lda, int ldb, int ldc, int flags) {
  launch_gemm(st, A, Bt, bias, resid, nullptr, nullptr, C, M, N, K,
              lda, ldb, ldc, flags, 1, 0, 0, 0, 0, 0);
}

extern "C" void kernel_launch(void* const* d_in, const int* in_sizes, int n_in,
                              void* d_out, int out_size, void* d_ws, size_t ws_size,
                              hipStream_t stream) {
  (void)in_sizes; (void)n_in; (void)out_size; (void)ws_size;
#define PF(i) ((const float*)d_in[(i)])
#define PI(i) ((const int*)d_in[(i)])

  // ---- workspace bump allocator (64KB slack per buffer: GEMM A/B tile rows
  //      up to the 16-row tile edge and speculative prefetch may over-read) --
  char* wsp = (char*)d_ws;
  auto alloc = [&](size_t nfloats) -> float* {
    float* p = (float*)wsp;
    wsp += (nfloats * sizeof(float) + 65536 + 255) & ~(size_t)255;
    return p;
  };

  float* qe   = alloc((size_t)BB * QL * EPAD);
  float* pe   = alloc((size_t)BB * PL * EPAD);
  float* ae   = alloc((size_t)BB * NANS * ALEN * EPAD);
  float* hq   = alloc((size_t)BB * QL * HH);
  float* hp   = alloc((size_t)BB * PL * HH);
  float* ha   = alloc((size_t)BB * NANS * ALEN * HH);
  float* hqT  = alloc((size_t)BB * HH * QL);
  float* hpT  = alloc((size_t)BB * HH * PL);
  float* TA   = alloc((size_t)1024 * 512);
  float* TB   = alloc((size_t)1024 * 512);
  float* TC   = alloc((size_t)BB * QL * HH);
  float* c1   = alloc((size_t)BB * PL * HH);
  float* c2   = alloc((size_t)BB * QL * HH);
  float* hpWb = alloc((size_t)BB * PL * HH);
  float* hqW  = alloc((size_t)BB * QL * HH);
  float* sbuf = alloc((size_t)BB * QL * PL);
  float* simb = alloc((size_t)BB * QL * PL);
  float* S1   = alloc((size_t)BB * QL * PL);
  float* S2   = alloc((size_t)BB * QL * PL);
  float* Rb   = alloc((size_t)BB * QL * QL);
  float* t1v  = alloc((size_t)BB * PL);
  float* t2v  = alloc((size_t)BB * QL);
  float* agg  = alloc((size_t)BB * QL * D8);
  float* aE   = alloc((size_t)BB * NANS * ALEN * EE);
  float* ascr = alloc((size_t)BB * NANS * ALEN);
  float* aemb = alloc((size_t)BB * NANS * EE);
  float* sj   = alloc((size_t)BB * QL);
  float* rq   = alloc((size_t)BB * HH);
  float* rqW  = alloc((size_t)BB * HH);
  float* s2b  = alloc((size_t)BB * QL);
  float* rp   = alloc((size_t)BB * D8);
  float* eo   = alloc((size_t)BB * EE);

  // ---- stage all weights transposed (N-major, zero-padded) -----------------
  auto tpose = [&](int pidx, int rows, int cols, int drows, int dcols) -> float* {
    float* d = alloc((size_t)drows * dcols);
    transpose_pad_kernel<<<drows, 256, 0, stream>>>(PF(pidx), d, rows, cols, dcols);
    return d;
  };
  float* wPQT  = tpose(P_PROJ_Q_W, EE, HH, HH, EPAD);
  float* wPPT  = tpose(P_PROJ_P_W, EE, HH, HH, EPAD);
  float* wPAT  = tpose(P_PROJ_A_W, EE, HH, HH, EPAD);
  float* wAPT  = tpose(P_A_PROJ_W, HH, EE, EPAD16, HH);
  float* wPRDT = tpose(P_PRED, D8, EE, EPAD16, D8);
  float* convPwT[2], *aconvPwT[2], *encW1T[3], *encW2T[3], *aencW1T[4], *aencW2T[4];
  for (int l = 0; l < 2; ++l) {
    convPwT[l]  = tpose(P_CONV0 + l * 6 + 4, HH, HH, HH, HH);
    aconvPwT[l] = tpose(P_ACONV0 + l * 6 + 4, D8, D8, D8, D8);
  }
  for (int l = 0; l < 3; ++l) {
    encW1T[l] = tpose(P_ENC0 + l * 8 + 4, HH, FF, FF, HH);
    encW2T[l] = tpose(P_ENC0 + l * 8 + 6, FF, HH, HH, FF);
  }
  for (int l = 0; l < 4; ++l) {
    aencW1T[l] = tpose(P_AENC0 + l * 8 + 4, D8, FF, FF, D8);
    aencW2T[l] = tpose(P_AENC0 + l * 8 + 6, FF, D8, D8, FF);
  }
  float* WtC1  = tpose(P_WC1, HH, HH, HH, HH);
  float* WtC2  = tpose(P_WC2, HH, HH, HH, HH);
  float* WtB   = tpose(P_WB, HH, HH, HH, HH);
  float* WtD   = tpose(P_WD, HH, HH, HH, HH);
  float* WtM   = tpose(P_WM, HH, HH, HH, HH);
  float* WtS   = tpose(P_WS, HH, HH, HH, HH);
  float* WtQA3 = tpose(P_WQA3, HH, HH, HH, HH);
  float* WtQ   = tpose(P_WQ, HH, HH, HH, HH);
  float* WtP1  = tpose(P_WP1, D8, HH, HH, D8);
  float* WtP2  = tpose(P_WP2, HH, HH, HH, HH);

  // ---- shared pre-norm residual sublayer helpers ---------------------------
  auto run_conv_block = [&](float* x, int Bb, int L, int C, int base,
                            float* const* pwT, int lnG, int lnB) {
    int rows = Bb * L;
    for (int layer = 0; layer < 2; ++layer) {
      int pi = base + layer * 6;  // ln_g, ln_b, dw, dw_b, pw, pw_b
      layernorm_kernel<<<rows, 256, 0, stream>>>(x, PF(pi), PF(pi + 1), TA, C);
      dwconv_kernel<<<dim3(L, Bb), 256, 0, stream>>>(TA, PF(pi + 2), PF(pi + 3), TB, L, C);
      gemm1(stream, TB, pwT[layer], PF(pi + 5), x, x, rows, C, C, C, C, C,
            EP_BIAS | EP_RELU | EP_RESID);
    }
    layernorm_kernel<<<rows, 256, 0, stream>>>(x, PF(lnG), PF(lnB), x, C);
  };

  auto run_encoder = [&](float* x, int Bb, int L, int D, const int* tokens,
                         int base, int nLayers, float* const* w1T,
                         float* const* w2T, int lnG, int lnB) {
    int rows = Bb * L;
    for (int l = 0; l < nLayers; ++l) {
      int pi = base + l * 8;  // ln1_g, ln1_b, ln2_g, ln2_b, w1, b1, w2, b2
      layernorm_kernel<<<rows, 256, 0, stream>>>(x, PF(pi), PF(pi + 1), TA, D);
      mha_kernel<<<dim3(L, Bb), 256, 0, stream>>>(TA, x, tokens, x, L, D, NHEADS);
      layernorm_kernel<<<rows, 256, 0, stream>>>(x, PF(pi + 2), PF(pi + 3), TA, D);
      gemm1(stream, TA, w1T[l], PF(pi + 5), nullptr, TB, rows, FF, D, D, D, FF,
            EP_BIAS | EP_RELU);
      gemm1(stream, TB, w2T[l], PF(pi + 7), x, x, rows, D, FF, FF, FF, D,
            EP_BIAS | EP_RESID);
    }
    layernorm_kernel<<<rows, 256, 0, stream>>>(x, PF(lnG), PF(lnB), x, D);
  };

  // ---- embed + posenc + input projections ----------------------------------
  embed_posenc_kernel<<<BB * QL, 256, 0, stream>>>(PI(IN_QUERY), PF(P_EMB), qe, QL, EE, EPAD);
  embed_posenc_kernel<<<BB * PL, 256, 0, stream>>>(PI(IN_PASSAGE), PF(P_EMB), pe, PL, EE, EPAD);
  embed_posenc_kernel<<<BB * NANS * ALEN, 256, 0, stream>>>(PI(IN_ANSWER), PF(P_EMB), ae, ALEN, EE, EPAD);

  gemm1(stream, qe, wPQT, PF(P_PROJ_Q_B), nullptr, hq, BB * QL, HH, EPAD, EPAD, EPAD, HH, EP_BIAS);
  gemm1(stream, pe, wPPT, PF(P_PROJ_P_B), nullptr, hp, BB * PL, HH, EPAD, EPAD, EPAD, HH, EP_BIAS);
  gemm1(stream, ae, wPAT, PF(P_PROJ_A_B), nullptr, ha, BB * NANS * ALEN, HH, EPAD, EPAD, EPAD, HH, EP_BIAS);

  // ---- answer branch: conv + encoder + pooled embedding --------------------
  run_conv_block(ha, BB * NANS, ALEN, HH, P_CONV0, convPwT, P_CONV_LN_G, P_CONV_LN_B);
  run_encoder(ha, BB * NANS, ALEN, HH, PI(IN_ANSWER), P_ENC0, 3, encW1T, encW2T,
              P_ENC_LN_G, P_ENC_LN_B);
  gemm1(stream, ha, wAPT, PF(P_A_PROJ_B), nullptr, aE,
        BB * NANS * ALEN, EE, HH, HH, HH, EE, EP_BIAS);
  rowdot_kernel<<<BB * NANS * ALEN, 256, 0, stream>>>(aE, PF(P_A_ATT), ascr, EE, EE, 0);
  pool_rows_kernel<<<BB * NANS, 256, 0, stream>>>(ascr, aE, aemb, ALEN, EE);

  // ---- passage / query encoders --------------------------------------------
  run_conv_block(hp, BB, PL, HH, P_CONV0, convPwT, P_CONV_LN_G, P_CONV_LN_B);
  run_encoder(hp, BB, PL, HH, PI(IN_PASSAGE), P_ENC0, 3, encW1T, encW2T,
              P_ENC_LN_G, P_ENC_LN_B);
  run_conv_block(hq, BB, QL, HH, P_CONV0, convPwT, P_CONV_LN_G, P_CONV_LN_B);
  run_encoder(hq, BB, QL, HH, PI(IN_QUERY), P_ENC0, 3, encW1T, encW2T,
              P_ENC_LN_G, P_ENC_LN_B);
  // transposed copies of hp/hq for N-major B operands in p2q / q2p
  transpose_batch_kernel<<<dim3(HH, BB), 256, 0, stream>>>(hp, hpT, PL, HH);
  transpose_batch_kernel<<<dim3(HH, BB), 256, 0, stream>>>(hq, hqT, QL, HH);

  const long sQH = (long)QL * HH, sPH = (long)PL * HH;
  const long sQP = (long)QL * PL, sAgg = (long)QL * D8;

  // ---- concat attention (Wc1/Wc2/vc) -> agg slice 2 ------------------------
  gemm1(stream, hp, WtC1, nullptr, nullptr, c1, BB * PL, HH, HH, HH, HH, HH, 0);
  gemm1(stream, hq, WtC2, nullptr, nullptr, c2, BB * QL, HH, HH, HH, HH, HH, 0);
  addtanh_score_kernel<<<dim3(QL, BB), 256, 0, stream>>>(c1, c2, PF(P_VC), sbuf, QL, PL, HH);
  attn_pool_kernel<<<dim3(QL, BB), 256, 0, stream>>>(sbuf, hp, agg + 2 * HH, QL, PL, HH, D8, sAgg);

  // ---- bilinear attention (Wb) -> agg slice 4 ------------------------------
  gemm1(stream, hp, WtB, nullptr, nullptr, hpWb, BB * PL, HH, HH, HH, HH, HH, 0);
  launch_gemm(stream, hq, hpWb, nullptr, nullptr, nullptr, nullptr, sbuf,
              QL, PL, HH, HH, HH, PL, 0, BB, sQH, sPH, sQP, 0, 0);
  attn_pool_kernel<<<dim3(QL, BB), 256, 0, stream>>>(sbuf, hp, agg + 4 * HH, QL, PL, HH, D8, sAgg);

  // ---- dot attention (Wd) -> agg slice 3 (fused, no rank-4 materialization)-
  mw_tanh_score_kernel<<<dim3(PL / 16, QL, BB), 32, 0, stream>>>(
      hp, hq, WtD, PF(P_VD), sbuf, PL, QL, HH, 0);
  attn_pool_kernel<<<dim3(QL, BB), 256, 0, stream>>>(sbuf, hp, agg + 3 * HH, QL, PL, HH, D8, sAgg);

  // ---- minus attention (Wm) -> agg slice 5 ---------------------------------
  mw_tanh_score_kernel<<<dim3(PL / 16, QL, BB), 32, 0, stream>>>(
      hp, hq, WtM, PF(P_VM), sbuf, PL, QL, HH, 1);
  attn_pool_kernel<<<dim3(QL, BB), 256, 0, stream>>>(sbuf, hp, agg + 5 * HH, QL, PL, HH, D8, sAgg);

  // ---- self dot attention (Ws, hq x hq) -> agg slice 1 ---------------------
  mw_tanh_score_kernel<<<dim3(QL / 16, QL, BB), 32, 0, stream>>>(
      hq, hq, WtS, PF(P_VS), sbuf, QL, QL, HH, 0);
  attn_pool_kernel<<<dim3(QL, BB), 256, 0, stream>>>(sbuf, hq, agg + 1 * HH, QL, QL, HH, D8, sAgg);

  // ---- trilinear similarity + bidirectional attention ----------------------
  rowdot_kernel<<<BB * PL, 256, 0, stream>>>(hp, PF(P_WQA1), t1v, HH, HH, 0);
  rowdot_kernel<<<BB * QL, 256, 0, stream>>>(hq, PF(P_WQA2), t2v, HH, HH, 0);
  gemm1(stream, hq, WtQA3, nullptr, nullptr, hqW, BB * QL, HH, HH, HH, HH, HH, 0);
  launch_gemm(stream, hqW, hp, nullptr, nullptr, t1v, t2v, simb,
              QL, PL, HH, HH, HH, PL, EP_ADDROW | EP_ADDCOL,
              BB, sQH, sPH, sQP, PL, QL);
  softmax_lastdim_kernel<<<BB * QL, 256, 0, stream>>>(S1, simb, PL);
  softmax_dim1_kernel<<<dim3(PL, BB), 64, 0, stream>>>(S2, simb, QL, PL);
  // p2q = S1 @ hp -> agg slice 6 (B = hp^T, N-major)
  launch_gemm(stream, S1, hpT, nullptr, nullptr, nullptr, nullptr, agg + 6 * HH,
              QL, HH, PL, PL, PL, D8, 0, BB, sQP, (long)HH * PL, sAgg, 0, 0);
  // R = S1 @ S2^T ; q2p = R @ hq -> agg slice 7 (B = hq^T, N-major)
  launch_gemm(stream, S1, S2, nullptr, nullptr, nullptr, nullptr, Rb,
              QL, QL, PL, PL, PL, QL, 0, BB, sQP, sQP, (long)QL * QL, 0, 0);
  launch_gemm(stream, Rb, hqT, nullptr, nullptr, nullptr, nullptr, agg + 7 * HH,
              QL, HH, QL, QL, QL, D8, 0, BB, (long)QL * QL, (long)HH * QL, sAgg, 0, 0);
  // hq -> agg slice 0
  copy_slice_kernel<<<BB * QL, 256, 0, stream>>>(hq, agg, HH, HH, D8);

  // ---- aggregation conv + encoder ------------------------------------------
  run_conv_block(agg, BB, QL, D8, P_ACONV0, aconvPwT, P_ACONV_LN_G, P_ACONV_LN_B);
  run_encoder(agg, BB, QL, D8, PI(IN_QUERY), P_AENC0, 4, aencW1T, aencW2T,
              P_AENC_LN_G, P_AENC_LN_B);

  // ---- prediction layer ----------------------------------------------------
  gemm1(stream, hq, WtQ, nullptr, nullptr, TA, BB * QL, HH, HH, HH, HH, HH, 0);
  rowdot_kernel<<<BB * QL, 256, 0, stream>>>(TA, PF(P_VQ), sj, HH, HH, 1);
  pool_rows_kernel<<<BB, 256, 0, stream>>>(sj, hq, rq, QL, HH);
  gemm1(stream, rq, WtP2, nullptr, nullptr, rqW, BB, HH, HH, HH, HH, HH, 0);
  launch_gemm(stream, agg, WtP1, nullptr, nullptr, rqW, nullptr, TC,
              QL, HH, D8, D8, D8, HH, EP_ADDROW, BB, sAgg, 0, sQH, HH, 0);
  rowdot_kernel<<<BB * QL, 256, 0, stream>>>(TC, PF(P_VP), s2b, HH, HH, 0);
  pool_rows_kernel<<<BB, 256, 0, stream>>>(s2b, agg, rp, QL, D8);
  gemm1(stream, rp, wPRDT, nullptr, nullptr, eo, BB, EE, D8, D8, D8, EE, EP_LEAKY);

  // ---- loss ----------------------------------------------------------------
  zero_out_kernel<<<1, 1, 0, stream>>>((float*)d_out);
  loss_kernel<<<BB, 256, 0, stream>>>(aemb, eo, (float*)d_out, NANS, EE, BB);
#undef PF
#undef PI
}
